// JCA_11081015624162
// MI455X (gfx1250) — matrix operand order repaired and checked by reading, not compiled, not run on hardware
//
#include <hip/hip_runtime.h>
#include <hip/hip_bf16.h>
#include <math.h>

// ---------------------------------------------------------------------------
// MI455X / gfx1250 fused JCA block.
//   Kernel 0: one-shot fp32 -> bf16 conversion of GEMM operands (bandwidth
//             trivial: ~90MB total at 23.3 TB/s). Removes all cvt VALU from
//             the WMMA hot loops.
//   Kernel 1: encoder GEMMs (f @ We^T + be) -> aud/vis fp32 + av bf16 (ws)
//   Kernel 2: per (batch,stream) fused  tanh(tmp^T av)·W_c^T + feat·W_a^T
//             -> relu -> ·W_h^T + residual, all WMMA bf16 / f32-accum.
// WMMA operand layouts per CDNA5 ISA 7.12.2:
//   A (16x32 bf16): lane L holds row M=L%16; V0..3 = K (L<16?0:8)+0..7,
//                   V4..7 = K 16+(L<16?0:8)+0..7  (2 bf16 / VGPR)
//   B fed as Bt[n][k] (= B^T row-major): lane L holds col N=L%16;
//                   same K order as A; tmp·av and tanh·Wc use K (L<16?0:16)+0..15 for A and B
//   C/D (16x16 f32): lane L holds col N=L%16, rows M = v + 8*(L/16)
// ---------------------------------------------------------------------------

typedef __attribute__((ext_vector_type(16))) __bf16        v16bf;
typedef __attribute__((ext_vector_type(2)))  __bf16        v2bf;
typedef __attribute__((ext_vector_type(8)))  float         v8f;
typedef __attribute__((ext_vector_type(4)))  unsigned int  u32x4;
typedef __attribute__((ext_vector_type(4)))  float         f32x4;

union Frag { v16bf v; unsigned int u[8]; unsigned short us[16]; };

#define WMMA_BF16(A, B, C) \
  __builtin_amdgcn_wmma_f32_16x16x32_bf16(false, (A), false, (B), (short)0, (C), false, false)

__device__ __forceinline__ unsigned int packbf(float lo, float hi) {
#if __has_builtin(__builtin_amdgcn_cvt_pk_bf16_f32)
  v2bf p = __builtin_amdgcn_cvt_pk_bf16_f32(lo, hi);
  return __builtin_bit_cast(unsigned int, p);
#else
  unsigned int a = __builtin_bit_cast(unsigned int, lo);
  unsigned int b = __builtin_bit_cast(unsigned int, hi);
  a += 0x7FFFu + ((a >> 16) & 1u);
  b += 0x7FFFu + ((b >> 16) & 1u);
  return (a >> 16) | (b & 0xFFFF0000u);
#endif
}
__device__ __forceinline__ unsigned short f2bf(float x) {
  return (unsigned short)packbf(x, x);
}
__device__ __forceinline__ float fast_tanh(float x) {
#if __has_builtin(__builtin_amdgcn_tanhf)
  return __builtin_amdgcn_tanhf(x);
#elif __has_builtin(__builtin_amdgcn_tanh_f32)
  return __builtin_amdgcn_tanh_f32(x);
#else
  return tanhf(x);
#endif
}

// ==================== Kernel 0: fp32 -> bf16 operand prepack ===============
__global__ __launch_bounds__(256) void jca_cvt(const float* __restrict__ src,
                                               unsigned int* __restrict__ dst,
                                               int npair) {
  int i = blockIdx.x * 256 + threadIdx.x;     // 4 packed pairs (8 floats) each
  int idx = i * 4;
  if (idx + 3 < npair) {
    f32x4 x0 = *(const f32x4*)(src + (size_t)idx * 2);
    f32x4 x1 = *(const f32x4*)(src + (size_t)idx * 2 + 4);
    u32x4 o;
    o[0] = packbf(x0[0], x0[1]); o[1] = packbf(x0[2], x0[3]);
    o[2] = packbf(x1[0], x1[1]); o[3] = packbf(x1[2], x1[3]);
    *(u32x4*)(dst + idx) = o;
  }
}

// ======================= Kernel 1: encoder GEMM ============================
// 2 streams x (M=12800, N=384, K=768), bf16 operands. One wave -> 64x32 block
// (4x2 tiles): pure b128 loads + 8 WMMA per k-step, no conversion VALU.
__global__ __launch_bounds__(128) void jca_enc(
    const unsigned short* __restrict__ f1b, const unsigned short* __restrict__ f2b,
    const unsigned short* __restrict__ We1b, const float* __restrict__ be1,
    const unsigned short* __restrict__ We2b, const float* __restrict__ be2,
    float* __restrict__ ws_feat, unsigned short* __restrict__ ws_av)
{
  const int lane = threadIdx.x & 31;
  const int wave = threadIdx.x >> 5;
  const int rl = lane & 15, hi = lane >> 4;

  int tile = blockIdx.x * 4 + wave;            // 0 .. 4799
  int st = tile / (200 * 12);
  int r  = tile % (200 * 12);
  int mb = r / 12, nb = r % 12;                // 64-row m-block, 32-col n-block

  const unsigned short* F  = st ? f2b  : f1b;
  const unsigned short* We = st ? We2b : We1b;
  const float*          be = st ? be2  : be1;

  const int g0 = mb * 64;
  const int n0 = nb * 32;
  const unsigned short* arow = F  + (size_t)(g0 + rl) * 768;
  const unsigned short* brow = We + (size_t)(n0 + rl) * 768;

  v8f c[4][2] = {};
  for (int kk = 0; kk < 24; ++kk) {
    Frag bfr[2];
#pragma unroll
    for (int jn = 0; jn < 2; ++jn) {
      const unsigned short* bp = brow + (size_t)jn * 16 * 768 + kk * 32 + hi * 8;
      ((u32x4*)bfr[jn].u)[0] = *(const u32x4*)bp;
      ((u32x4*)bfr[jn].u)[1] = *(const u32x4*)(bp + 16);
    }
#pragma unroll
    for (int jm = 0; jm < 4; ++jm) {
      Frag a;
      const unsigned short* ap = arow + (size_t)jm * 16 * 768 + kk * 32;
      ((u32x4*)a.u)[0] = *(const u32x4*)(ap + hi * 8);
      ((u32x4*)a.u)[1] = *(const u32x4*)(ap + 16 + hi * 8);
      c[jm][0] = WMMA_BF16(a.v, bfr[0].v, c[jm][0]);
      c[jm][1] = WMMA_BF16(a.v, bfr[1].v, c[jm][1]);
    }
  }

#pragma unroll
  for (int jm = 0; jm < 4; ++jm) {
#pragma unroll
    for (int jn = 0; jn < 2; ++jn) {
      const int hcol = n0 + jn * 16 + rl;      // C col = lane%16
      const float bias = be[hcol];
#pragma unroll
      for (int v = 0; v < 8; ++v) {
        int g = g0 + jm * 16 + v + 8 * hi;     // C row = v + 8*hi
        float val = c[jm][jn][v] + bias;
        ws_feat[(size_t)st * 12800 * 384 + (size_t)g * 384 + hcol] = val;
        // (b*25 + s) == g, so no div/mod needed:
        ws_av[(size_t)g * 768 + st * 384 + hcol] = f2bf(val);
      }
    }
  }
}

// ================= Kernel 2: fused attention per (batch,stream) ============
__global__ __launch_bounds__(256) void jca_att(
    const float* __restrict__ Waff_a, const float* __restrict__ Waff_v,
    const float* __restrict__ W_a,  const float* __restrict__ W_v,
    const unsigned short* __restrict__ Wcab, const unsigned short* __restrict__ Wcvb,
    const float* __restrict__ W_ha, const float* __restrict__ W_hv,
    const float* __restrict__ ws_feat, const unsigned short* __restrict__ ws_av,
    float* __restrict__ out)
{
  extern __shared__ char smem[];
  float*          sFeat  = (float*)(smem);                     // [25][384] f32   38400
  float*          sWaff  = (float*)(smem + 38400);             // [25][25]  f32    2560(pad)
  unsigned short* sAvT   = (unsigned short*)(smem + 40960);    // [768][32] bf16  49152
  unsigned short* sTmpA  = (unsigned short*)(smem + 90112);    // [384][32] bf16  24576
  unsigned short* sFeatT = (unsigned short*)(smem + 114688);   // [384][32] bf16  24576
  unsigned short* sStage = (unsigned short*)(smem + 139264);   // 8*[16][32]       8192
  unsigned short* sH     = (unsigned short*)(smem + 147456);   // [384][128] bf16 98304
                                                               // total 245760 B

  const int tid = threadIdx.x;
  const int lane = tid & 31, wave = tid >> 5;
  const int rl = lane & 15, hi = lane >> 4;
  const int b = blockIdx.x >> 1, st = blockIdx.x & 1;

  const float* Waff = st ? Waff_v : Waff_a;
  const float* Wsm  = st ? W_v  : W_a;
  const unsigned short* Wcb = st ? Wcvb : Wcab;
  const float* Wh   = st ? W_hv : W_ha;
  const float* gFeat = ws_feat + ((size_t)st * 512 + b) * 25 * 384;
  const float scale = 0.03608439182435161f;    // 1/sqrt(768)

  // ---- Phase 0: stage LDS -------------------------------------------------
  for (int i = tid; i < 25 * 384; i += 256) sFeat[i] = gFeat[i];
  for (int i = tid; i < 625;      i += 256) sWaff[i] = Waff[i];
  for (int i = tid; i < 768 * 32; i += 256) {
    int e = i >> 5, s = i & 31;
    sAvT[i] = (s < 25) ? ws_av[((size_t)b * 25 + s) * 768 + e] : (unsigned short)0;
  }
  for (int i = tid; i < 384 * 128; i += 256) sH[i] = 0;
  __syncthreads();
  // tmp[s,d] = sum_p feat[p,d]*Waff[p,s]; store scaled, transposed, padded
  for (int i = tid; i < 384 * 32; i += 256) {
    int d = i >> 5, s = i & 31;
    float acc = 0.f;
    if (s < 25) {
      for (int p = 0; p < 25; ++p) acc += sFeat[p * 384 + d] * sWaff[p * 25 + s];
      acc *= scale;
    }
    sTmpA[i]  = f2bf(acc);
    sFeatT[i] = (s < 25) ? f2bf(sFeat[s * 384 + d]) : (unsigned short)0;
  }
  __syncthreads();

  // ---- Phase 1: H[d,a] = relu( tanh(tmp^T av)·Wc^T + feat^T·Wsm^T ) -------
  Frag waB[7];                                 // B-frags of Wsm [100,25] (Bt = row-major)
#pragma unroll
  for (int at = 0; at < 7; ++at) {
    int a = at * 16 + rl;
#pragma unroll
    for (int i = 0; i < 16; ++i) {
      int k = (i < 8) ? hi * 8 + i : 16 + hi * 8 + (i - 8);
      waB[at].us[i] = (a < 100 && k < 25) ? f2bf(Wsm[a * 25 + k]) : (unsigned short)0;
    }
  }

  for (int dt = wave; dt < 24; dt += 8) {      // wave-uniform
    Frag aG, aT;
    const char* fr = (const char*)sFeatT + (dt * 16 + rl) * 64;
    ((u32x4*)aG.u)[0] = *(const u32x4*)(fr + hi * 16);
    ((u32x4*)aG.u)[1] = *(const u32x4*)(fr + 32 + hi * 16);
    const char* tr = (const char*)sTmpA + (dt * 16 + rl) * 64;
    ((u32x4*)aT.u)[0] = *(const u32x4*)(tr + hi * 32);
    ((u32x4*)aT.u)[1] = *(const u32x4*)(tr + hi * 32 + 16);

    v8f acc[7];
#pragma unroll
    for (int at = 0; at < 7; ++at) { v8f z = {}; acc[at] = WMMA_BF16(aG.v, waB[at].v, z); }

    unsigned short* stg = sStage + wave * 512; // wave-private 16x32 staging tile
    for (int et = 0; et < 24; ++et) {
      Frag b0, b1;
      const char* r0 = (const char*)sAvT + (et * 32 + rl) * 64;
      ((u32x4*)b0.u)[0] = *(const u32x4*)(r0 + hi * 32);
      ((u32x4*)b0.u)[1] = *(const u32x4*)(r0 + hi * 32 + 16);
      const char* r1 = r0 + 16 * 64;
      ((u32x4*)b1.u)[0] = *(const u32x4*)(r1 + hi * 32);
      ((u32x4*)b1.u)[1] = *(const u32x4*)(r1 + hi * 32 + 16);
      v8f z0 = {}, z1 = {};
      v8f t0 = WMMA_BF16(aT.v, b0.v, z0);
      v8f t1 = WMMA_BF16(aT.v, b1.v, z1);

      // tanh -> bf16 -> A-layout staging (wave-internal; LDS is in-order)
#pragma unroll
      for (int v = 0; v < 8; ++v) {
        int row = v + 8 * hi;
        stg[row * 32 + rl]      = f2bf(fast_tanh(t0[v]));
        stg[row * 32 + 16 + rl] = f2bf(fast_tanh(t1[v]));
      }
      asm volatile("" ::: "memory");
      Frag aA;
      const char* ar = (const char*)stg + rl * 64;
      ((u32x4*)aA.u)[0] = *(const u32x4*)(ar + hi * 32);
      ((u32x4*)aA.u)[1] = *(const u32x4*)(ar + hi * 32 + 16);
      asm volatile("" ::: "memory");

#pragma unroll
      for (int at = 0; at < 7; ++at) {
        Frag bc;
        int a = at * 16 + rl;
        if (a < 100) {
          const unsigned short* wr = Wcb + (size_t)a * 768 + et * 32 + hi * 16;
          ((u32x4*)bc.u)[0] = *(const u32x4*)wr;
          ((u32x4*)bc.u)[1] = *(const u32x4*)(wr + 8);
        } else {
#pragma unroll
          for (int j = 0; j < 8; ++j) bc.u[j] = 0;
        }
        acc[at] = WMMA_BF16(aA.v, bc.v, acc[at]);
      }
    }
    // relu -> bf16 H rows (row-major [d][128] == Bt layout for final GEMM)
#pragma unroll
    for (int at = 0; at < 7; ++at) {
      int a = at * 16 + rl;
#pragma unroll
      for (int v = 0; v < 8; ++v) {
        float x = acc[at][v]; x = x > 0.f ? x : 0.f;
        int d = dt * 16 + v + 8 * hi;
        sH[d * 128 + a] = f2bf(x);
      }
    }
  }
  __syncthreads();

  // ---- Phase 2: out[s,d] = sum_a Wh[s,a]*H[d,a] + feat[s,d] ---------------
  // Hoist all 8 possible A-fragments of Wh [25,100] (2 m-tiles x 4 k-steps).
  Frag aW[2][4];
#pragma unroll
  for (int mt = 0; mt < 2; ++mt) {
    int srow = mt * 16 + rl;
#pragma unroll
    for (int ks = 0; ks < 4; ++ks) {
#pragma unroll
      for (int i = 0; i < 8; ++i) {
        int k = ks * 32 + hi * 8 + i;
        aW[mt][ks].us[i] =
            (srow < 25 && k < 100) ? f2bf(Wh[srow * 100 + k]) : (unsigned short)0;
      }
#pragma unroll
      for (int i = 0; i < 8; ++i) {
        int k = ks * 32 + 16 + hi * 8 + i;
        aW[mt][ks].us[8 + i] =
            (srow < 25 && k < 100) ? f2bf(Wh[srow * 100 + k]) : (unsigned short)0;
      }
    }
  }

  for (int t = wave; t < 48; t += 8) {         // wave-uniform
    int mt = t & 1, nt = t >> 1;
    v8f c = {};
#pragma unroll
    for (int ks = 0; ks < 4; ++ks) {
      Frag bH;
      const char* hr = (const char*)sH + (nt * 16 + rl) * 256 + ks * 64 + hi * 16;
      ((u32x4*)bH.u)[0] = *(const u32x4*)hr;
      ((u32x4*)bH.u)[1] = *(const u32x4*)(hr + 32);
      c = WMMA_BF16(aW[mt][ks].v, bH.v, c);
    }
#pragma unroll
    for (int v = 0; v < 8; ++v) {
      int s = mt * 16 + v + 8 * hi;
      if (s < 25) {
        int d = nt * 16 + rl;
        out[((size_t)b * 25 + s) * 768 + st * 384 + d] = c[v] + sFeat[s * 384 + d];
      }
    }
  }
}

// ============================== launcher ===================================
extern "C" void kernel_launch(void* const* d_in, const int* in_sizes, int n_in,
                              void* d_out, int out_size, void* d_ws, size_t ws_size,
                              hipStream_t stream) {
  const float* f1     = (const float*)d_in[0];
  const float* f2     = (const float*)d_in[1];
  const float* We1    = (const float*)d_in[2];
  const float* be1    = (const float*)d_in[3];
  const float* We2    = (const float*)d_in[4];
  const float* be2    = (const float*)d_in[5];
  const float* Waff_a = (const float*)d_in[6];
  const float* Waff_v = (const float*)d_in[7];
  const float* W_a    = (const float*)d_in[8];
  const float* W_v    = (const float*)d_in[9];
  const float* W_ca   = (const float*)d_in[10];
  const float* W_cv   = (const float*)d_in[11];
  const float* W_ha   = (const float*)d_in[12];
  const float* W_hv   = (const float*)d_in[13];

  char* ws = (char*)d_ws;
  float*          ws_feat = (float*)(ws);                        // 39,321,600 B
  unsigned short* ws_av   = (unsigned short*)(ws + 39321600);    // 19,660,800 B
  unsigned short* f1b     = (unsigned short*)(ws + 58982400);    // 19,660,800 B
  unsigned short* f2b     = (unsigned short*)(ws + 78643200);    // 19,660,800 B
  unsigned short* We1b    = (unsigned short*)(ws + 98304000);    //    589,824 B
  unsigned short* We2b    = (unsigned short*)(ws + 98893824);    //    589,824 B
  unsigned short* Wcab    = (unsigned short*)(ws + 99483648);    //    153,600 B
  unsigned short* Wcvb    = (unsigned short*)(ws + 99637248);    //    153,600 B

  // Kernel 0: operand prepack (n/2 pairs each; all sizes divisible by 8)
  jca_cvt<<<4800, 256, 0, stream>>>(f1,  (unsigned int*)f1b,  9830400 / 2);
  jca_cvt<<<4800, 256, 0, stream>>>(f2,  (unsigned int*)f2b,  9830400 / 2);
  jca_cvt<<<144,  256, 0, stream>>>(We1, (unsigned int*)We1b,  294912 / 2);
  jca_cvt<<<144,  256, 0, stream>>>(We2, (unsigned int*)We2b,  294912 / 2);
  jca_cvt<<<38,   256, 0, stream>>>(W_ca,(unsigned int*)Wcab,   76800 / 2);
  jca_cvt<<<38,   256, 0, stream>>>(W_cv,(unsigned int*)Wcvb,   76800 / 2);

  jca_enc<<<1200, 128, 0, stream>>>(f1b, f2b, We1b, be1, We2b, be2, ws_feat, ws_av);
  jca_att<<<1024, 256, 245760, stream>>>(Waff_a, Waff_v, W_a, W_v, Wcab, Wcvb,
                                         W_ha, W_hv, ws_feat, ws_av, (float*)d_out);
}
